// CausalSelfAttention_44289702756581
// MI455X (gfx1250) — matrix operand order repaired and checked by reading, not compiled, not run on hardware
//
#include <hip/hip_runtime.h>

// ---------------------------------------------------------------------------
// CDNA5 (gfx1250, wave32) causal self-attention fwd:
//   1) convert x, w_qkv^T, w_out^T -> bf16
//   2) qkv = x @ w_qkv + b  (WMMA bf16, 32x64 register-blocked per wave)
//      -> Q[b,h,t,d], K[b,h,t,d], V^T[b,h,d,t]
//   3) flash attention, 16 q-rows/wave, 32-key chunks; K/V tiles staged
//      Global->LDS by the Tensor Data Mover (double-buffered, TENSORcnt)
//   4) out = y @ w_out + b  (WMMA bf16, f32 out)
// ---------------------------------------------------------------------------

typedef __attribute__((ext_vector_type(16))) __bf16        bf16x16;
typedef __attribute__((ext_vector_type(8)))  float         floatx8;
typedef __attribute__((ext_vector_type(4)))  unsigned int  uintx4;
typedef __attribute__((ext_vector_type(4)))  int           intx4;
typedef __attribute__((ext_vector_type(8)))  int           intx8;

union FragBF { bf16x16 f; uintx4 u[2]; };

__device__ __forceinline__ unsigned short f2bf(float x) {
  unsigned int u = __float_as_uint(x);
  u += 0x7FFFu + ((u >> 16) & 1u);          // round-to-nearest-even
  return (unsigned short)(u >> 16);
}

__device__ __forceinline__ floatx8 wmma_bf16(bf16x16 a, bf16x16 b, floatx8 c) {
  return __builtin_amdgcn_wmma_f32_16x16x32_bf16(false, a, false, b, (short)0, c,
                                                 false, false);
}

// ------------------------- Tensor Data Mover -------------------------------
#if defined(__has_builtin)
#if __has_builtin(__builtin_amdgcn_tensor_load_to_lds)
#define HAVE_TDM 1
#endif
#endif
#ifndef HAVE_TDM
#define HAVE_TDM 0
#endif

#if HAVE_TDM
// 2D tile load Global->LDS, bf16 elements. D# per CDNA5 ISA ch.8:
//  group0: count=1 | lds_addr | global_addr[56:0] | type=2
//  group1: data_size=2B, tensor_dim0/1, tile_dim0/1, tensor_dim0_stride
__device__ __forceinline__ void tdm_load_2d(unsigned lds_off, const void* gptr,
                                            unsigned tile_d0, unsigned tile_d1,
                                            unsigned stride_el, unsigned tdim0,
                                            unsigned tdim1) {
  const unsigned long long ga = (unsigned long long)(size_t)gptr;
  uintx4 g0;
  g0.x = 1u;                                                  // count=1
  g0.y = lds_off;                                             // lds_addr
  g0.z = (unsigned)ga;                                        // addr[31:0]
  g0.w = (unsigned)((ga >> 32) & 0x01FFFFFFull) | (2u << 30); // addr[56:32]|type=2
  intx8 g1;
  g1[0] = (int)(1u << 16);                                    // data_size=2B
  g1[1] = (int)((tdim0 & 0xFFFFu) << 16);                     // tensor_dim0 lo
  g1[2] = (int)(((tdim0 >> 16) & 0xFFFFu) | ((tdim1 & 0xFFFFu) << 16));
  g1[3] = (int)(((tdim1 >> 16) & 0xFFFFu) | ((tile_d0 & 0xFFFFu) << 16));
  g1[4] = (int)(tile_d1 & 0xFFFFu);                           // tile_dim2=0
  g1[5] = (int)stride_el;                                     // dim0_stride lo32
  g1[6] = 0;
  g1[7] = 0;
  const intx4 z4 = {0, 0, 0, 0};
#if __clang_major__ >= 23
  const intx8 z8 = {0, 0, 0, 0, 0, 0, 0, 0};
  __builtin_amdgcn_tensor_load_to_lds(g0, g1, z4, z4, z8, 0);
#else
  __builtin_amdgcn_tensor_load_to_lds(g0, g1, z4, z4, 0);
#endif
}
#endif

// --------------- 32x64 register-blocked WMMA GEMM tile ---------------------
// A: tile base (row m0), row-major stride K. Bt: transposed B, tile base
// (row n0), stride K. Produces 8 accumulators acc[nt*2+mt] = C(mt,nt).
__device__ __forceinline__ void gemm_tile_32x64_bf16(
    const unsigned short* __restrict__ A, const unsigned short* __restrict__ Bt,
    const int K, const int c, const int hi, floatx8 acc[8]) {
  const unsigned short* a0 = A + (size_t)c * K;
  const unsigned short* a1 = A + (size_t)(16 + c) * K;
  const unsigned short* b0 = Bt + (size_t)c * K;
  for (int k0 = 0; k0 < K; k0 += 32) {
    __builtin_prefetch(a0 + k0 + 256, 0, 0);
    __builtin_prefetch(b0 + k0 + 256, 0, 0);
    FragBF A0, A1;
    A0.u[0] = *(const uintx4*)(a0 + k0 + 8 * hi);
    A0.u[1] = *(const uintx4*)(a0 + k0 + 8 * hi + 16);
    A1.u[0] = *(const uintx4*)(a1 + k0 + 8 * hi);
    A1.u[1] = *(const uintx4*)(a1 + k0 + 8 * hi + 16);
#pragma unroll
    for (int nt = 0; nt < 4; ++nt) {
      const unsigned short* bp = b0 + (size_t)nt * 16 * K + k0 + 8 * hi;
      FragBF Bf;
      Bf.u[0] = *(const uintx4*)(bp);
      Bf.u[1] = *(const uintx4*)(bp + 16);
      acc[nt * 2 + 0] = wmma_bf16(A0.f, Bf.f, acc[nt * 2 + 0]);
      acc[nt * 2 + 1] = wmma_bf16(A1.f, Bf.f, acc[nt * 2 + 1]);
    }
  }
}

// ------------------------------- converts ----------------------------------
__global__ __launch_bounds__(256) void conv_bf16(const float* __restrict__ src,
                                                 unsigned short* __restrict__ dst,
                                                 int n) {
  int i = blockIdx.x * 256 + threadIdx.x;
  if (i < n) dst[i] = f2bf(src[i]);
}

// dst[N][K] = bf16(src[K][N])
__global__ __launch_bounds__(256) void conv_bf16_T(const float* __restrict__ src,
                                                   unsigned short* __restrict__ dst,
                                                   int K, int N) {
  int i = blockIdx.x * 256 + threadIdx.x;
  if (i < K * N) {
    int n = i / K, k = i - n * K;
    dst[i] = f2bf(src[(size_t)k * N + n]);
  }
}

// ------------------------------- QKV GEMM ----------------------------------
__global__ __launch_bounds__(128) void qkv_gemm(
    const unsigned short* __restrict__ Xb, const unsigned short* __restrict__ WqT,
    const float* __restrict__ bqkv, unsigned short* __restrict__ Qb,
    unsigned short* __restrict__ Kbuf, unsigned short* __restrict__ Vt) {
  const int lane = threadIdx.x & 31, wave = threadIdx.x >> 5;
  const int c = lane & 15, hi = lane >> 4;
  const int m0 = blockIdx.x << 5;                       // 32-row tile
  const int n0 = (blockIdx.y * 4 + wave) << 6;          // 64-col tile

  floatx8 acc[8] = {};
  gemm_tile_32x64_bf16(Xb + (size_t)m0 * 1024, WqT + (size_t)n0 * 1024, 1024, c,
                       hi, acc);
#pragma unroll
  for (int nt = 0; nt < 4; ++nt) {
    const int col = n0 + nt * 16 + c;
    const float bias = bqkv[col];
    const int sel = col >> 10;                          // 0=Q 1=K 2=V
    const int cc = col & 1023, hh = cc >> 6, d = cc & 63;
#pragma unroll
    for (int mt = 0; mt < 2; ++mt) {
#pragma unroll
      for (int r = 0; r < 8; ++r) {
        const int row = m0 + mt * 16 + r + 8 * hi;      // b*T + t
        const int bb = row >> 11, t = row & 2047;
        const unsigned short v = f2bf(acc[nt * 2 + mt][r] + bias);
        const size_t bhh = (size_t)(bb * 16 + hh);
        if (sel == 0)      Qb[(bhh * 2048 + t) * 64 + d] = v;
        else if (sel == 1) Kbuf[(bhh * 2048 + t) * 64 + d] = v;
        else               Vt[(bhh * 64 + d) * 2048 + t] = v;
      }
    }
  }
}

// --------------------------- flash attention -------------------------------
__global__ __launch_bounds__(32) void attn_fa(
    const unsigned short* __restrict__ Qb, const unsigned short* __restrict__ Kb,
    const unsigned short* __restrict__ Vt, unsigned short* __restrict__ Yb) {
  __shared__ __align__(16) unsigned short Ktile[2][32 * 64];  // [key][d]
  __shared__ __align__(16) unsigned short Vtile[2][64 * 32];  // [d][key]
  __shared__ __align__(16) unsigned short Plds[16 * 32];
  const int lane = threadIdx.x & 31;
  const int c = lane & 15, hi = lane >> 4;
  const int bh = blockIdx.x;                  // b*16 + h
  const int qbase = blockIdx.y << 4;
  const int b = bh >> 4, h = bh & 15;

  const unsigned short* Kbase = Kb + (size_t)bh * 2048 * 64;
  const unsigned short* Vbase = Vt + (size_t)bh * 64 * 2048;

  // Q fragments for d-chunks {0..31},{32..63} (reused across all key chunks)
  const unsigned short* Qrow = Qb + ((size_t)bh * 2048 + qbase + c) * 64;
  FragBF qf0, qf1;
  qf0.u[0] = *(const uintx4*)(Qrow + 16 * hi);
  qf0.u[1] = *(const uintx4*)(Qrow + 16 * hi + 8);
  qf1.u[0] = *(const uintx4*)(Qrow + 32 + 16 * hi);
  qf1.u[1] = *(const uintx4*)(Qrow + 32 + 16 * hi + 8);

  floatx8 o0 = {}, o1 = {}, o2 = {}, o3 = {};
  float m_i[8], l_i[8];
#pragma unroll
  for (int r = 0; r < 8; ++r) { m_i[r] = -1e30f; l_i[r] = 0.f; }

  const int kcEnd = (qbase + 15) >> 5;

#if HAVE_TDM
  const unsigned kof[2] = {(unsigned)(size_t)(void*)&Ktile[0][0],
                           (unsigned)(size_t)(void*)&Ktile[1][0]};
  const unsigned vof[2] = {(unsigned)(size_t)(void*)&Vtile[0][0],
                           (unsigned)(size_t)(void*)&Vtile[1][0]};
  // prologue: chunk 0 into buffer 0
  tdm_load_2d(kof[0], Kbase, 64, 32, 64, 64, 32);
  tdm_load_2d(vof[0], Vbase, 32, 64, 2048, 32, 64);
#endif

  for (int kc = 0; kc <= kcEnd; ++kc) {
    const int kb = kc << 5;
    const int buf = kc & 1;

#if HAVE_TDM
    if (kc < kcEnd) {
      // next chunk into other buffer; drain DS reads of that buffer first
      asm volatile("s_wait_dscnt 0" ::: "memory");
      tdm_load_2d(kof[buf ^ 1], Kbase + (size_t)(kb + 32) * 64, 64, 32, 64, 64, 32);
      tdm_load_2d(vof[buf ^ 1], Vbase + (kb + 32), 32, 64, 2048, 32, 64);
      __builtin_amdgcn_s_wait_tensorcnt(2);   // current buffer's pair done
    } else {
      __builtin_amdgcn_s_wait_tensorcnt(0);
    }
#else
    // fallback: cooperative wave copy Global->LDS
    {
      const unsigned short* gk = Kbase + (size_t)(kb + lane) * 64;
      unsigned short* lk = &Ktile[buf][lane * 64];
#pragma unroll
      for (int i = 0; i < 8; ++i)
        *(uintx4*)(lk + i * 8) = *(const uintx4*)(gk + i * 8);
#pragma unroll
      for (int rr = 0; rr < 2; ++rr) {
        const unsigned short* gv = Vbase + (size_t)(lane + 32 * rr) * 2048 + kb;
        unsigned short* lv = &Vtile[buf][(lane + 32 * rr) * 32];
#pragma unroll
        for (int i = 0; i < 4; ++i)
          *(uintx4*)(lv + i * 8) = *(const uintx4*)(gv + i * 8);
      }
      asm volatile("s_wait_dscnt 0" ::: "memory");
    }
#endif

    const unsigned short* Kt = &Ktile[buf][0];
    const unsigned short* Vp = &Vtile[buf][0];

    // S[16q x 32key] = Q @ K^T (fragments out of LDS)
    floatx8 s0 = {}, s1 = {};
    {
      const unsigned short* Kr0 = Kt + (size_t)c * 64;         // keys 0..15
      const unsigned short* Kr1 = Kt + (size_t)(16 + c) * 64;  // keys 16..31
      FragBF kf;
      kf.u[0] = *(const uintx4*)(Kr0 + 16 * hi);
      kf.u[1] = *(const uintx4*)(Kr0 + 16 * hi + 8);
      s0 = wmma_bf16(qf0.f, kf.f, s0);
      kf.u[0] = *(const uintx4*)(Kr0 + 32 + 16 * hi);
      kf.u[1] = *(const uintx4*)(Kr0 + 32 + 16 * hi + 8);
      s0 = wmma_bf16(qf1.f, kf.f, s0);
      kf.u[0] = *(const uintx4*)(Kr1 + 16 * hi);
      kf.u[1] = *(const uintx4*)(Kr1 + 16 * hi + 8);
      s1 = wmma_bf16(qf0.f, kf.f, s1);
      kf.u[0] = *(const uintx4*)(Kr1 + 32 + 16 * hi);
      kf.u[1] = *(const uintx4*)(Kr1 + 32 + 16 * hi + 8);
      s1 = wmma_bf16(qf1.f, kf.f, s1);
    }

    // scale, causal mask, online softmax (row = r + 8*hi, col across lanes)
#pragma unroll
    for (int r = 0; r < 8; ++r) {
      const int q = qbase + r + 8 * hi;
      float v0 = s0[r] * 0.125f;                       // 1/sqrt(64)
      float v1 = s1[r] * 0.125f;
      if (kb + c > q)      v0 = -1e30f;
      if (kb + 16 + c > q) v1 = -1e30f;
      float mx = fmaxf(v0, v1);
      mx = fmaxf(mx, __shfl_xor(mx, 1, 32));
      mx = fmaxf(mx, __shfl_xor(mx, 2, 32));
      mx = fmaxf(mx, __shfl_xor(mx, 4, 32));
      mx = fmaxf(mx, __shfl_xor(mx, 8, 32));
      const float mnew = fmaxf(m_i[r], mx);
      const float alpha = __expf(m_i[r] - mnew);
      const float p0 = __expf(v0 - mnew);
      const float p1 = __expf(v1 - mnew);
      float rs = p0 + p1;
      rs += __shfl_xor(rs, 1, 32);
      rs += __shfl_xor(rs, 2, 32);
      rs += __shfl_xor(rs, 4, 32);
      rs += __shfl_xor(rs, 8, 32);
      l_i[r] = l_i[r] * alpha + rs;
      m_i[r] = mnew;
      s0[r] = p0; s1[r] = p1;
      o0[r] *= alpha; o1[r] *= alpha; o2[r] *= alpha; o3[r] *= alpha;
    }

    // C-layout P -> LDS -> A-layout fragment (bf16)
#pragma unroll
    for (int r = 0; r < 8; ++r) {
      Plds[(r + 8 * hi) * 32 + c]      = f2bf(s0[r]);
      Plds[(r + 8 * hi) * 32 + 16 + c] = f2bf(s1[r]);
    }
    asm volatile("s_wait_dscnt 0" ::: "memory");
    FragBF pf;
    pf.u[0] = *(const uintx4*)(&Plds[c * 32 + 16 * hi]);
    pf.u[1] = *(const uintx4*)(&Plds[c * 32 + 16 * hi + 8]);

    // O += P @ V  (V tile in LDS is [d][key], rows contiguous in key)
    const unsigned short* Vr = Vp + (size_t)c * 32 + 16 * hi;
    FragBF vf;
    vf.u[0] = *(const uintx4*)(Vr);
    vf.u[1] = *(const uintx4*)(Vr + 8);
    o0 = wmma_bf16(pf.f, vf.f, o0);
    vf.u[0] = *(const uintx4*)(Vr + 512);
    vf.u[1] = *(const uintx4*)(Vr + 512 + 8);
    o1 = wmma_bf16(pf.f, vf.f, o1);
    vf.u[0] = *(const uintx4*)(Vr + 1024);
    vf.u[1] = *(const uintx4*)(Vr + 1024 + 8);
    o2 = wmma_bf16(pf.f, vf.f, o2);
    vf.u[0] = *(const uintx4*)(Vr + 1536);
    vf.u[1] = *(const uintx4*)(Vr + 1536 + 8);
    o3 = wmma_bf16(pf.f, vf.f, o3);
  }

  // y[b*T+t][h*64+d] = O / l
#pragma unroll
  for (int r = 0; r < 8; ++r) {
    const int row = b * 2048 + qbase + r + 8 * hi;
    const float inv = 1.0f / l_i[r];
    unsigned short* y = Yb + (size_t)row * 1024 + h * 64 + c;
    y[0]  = f2bf(o0[r] * inv);
    y[16] = f2bf(o1[r] * inv);
    y[32] = f2bf(o2[r] * inv);
    y[48] = f2bf(o3[r] * inv);
  }
}

// ------------------------------ out GEMM -----------------------------------
__global__ __launch_bounds__(128) void out_gemm(
    const unsigned short* __restrict__ Yb, const unsigned short* __restrict__ WoT,
    const float* __restrict__ bout, float* __restrict__ out) {
  const int lane = threadIdx.x & 31, wave = threadIdx.x >> 5;
  const int c = lane & 15, hi = lane >> 4;
  const int m0 = blockIdx.x << 5;
  const int n0 = (blockIdx.y * 4 + wave) << 6;

  floatx8 acc[8] = {};
  gemm_tile_32x64_bf16(Yb + (size_t)m0 * 1024, WoT + (size_t)n0 * 1024, 1024, c,
                       hi, acc);
#pragma unroll
  for (int nt = 0; nt < 4; ++nt) {
    const int col = n0 + nt * 16 + c;
    const float bias = bout[col];
#pragma unroll
    for (int mt = 0; mt < 2; ++mt) {
#pragma unroll
      for (int r = 0; r < 8; ++r) {
        const int row = m0 + mt * 16 + r + 8 * hi;
        out[(size_t)row * 1024 + col] = acc[nt * 2 + mt][r] + bias;
      }
    }
  }
}

// ------------------------------- launch ------------------------------------
extern "C" void kernel_launch(void* const* d_in, const int* in_sizes, int n_in,
                              void* d_out, int out_size, void* d_ws, size_t ws_size,
                              hipStream_t stream) {
  const float* x    = (const float*)d_in[0];   // [4,2048,1024]
  const float* wqkv = (const float*)d_in[1];   // [1024,3072]
  const float* bqkv = (const float*)d_in[2];   // [3072]
  const float* wout = (const float*)d_in[3];   // [1024,1024]
  const float* bout = (const float*)d_in[4];   // [1024]
  float* out = (float*)d_out;                  // [4,2048,1024] f32

  const int M = 8192, C = 1024;
  unsigned short* Xb  = (unsigned short*)d_ws;        // 8192*1024
  unsigned short* WqT = Xb + (size_t)M * C;           // 3072*1024 (w_qkv^T)
  unsigned short* WoT = WqT + (size_t)3072 * C;       // 1024*1024 (w_out^T)
  unsigned short* Qb  = WoT + (size_t)C * C;          // [64][2048][64]
  unsigned short* Kb  = Qb + (size_t)64 * 2048 * 64;
  unsigned short* Vt  = Kb + (size_t)64 * 2048 * 64;  // [64][64][2048]
  unsigned short* Yb  = Vt + (size_t)64 * 2048 * 64;  // 8192*1024

  conv_bf16<<<(M * C) / 256, 256, 0, stream>>>(x, Xb, M * C);
  conv_bf16_T<<<(3072 * C) / 256, 256, 0, stream>>>(wqkv, WqT, C, 3072);
  conv_bf16_T<<<(C * C) / 256, 256, 0, stream>>>(wout, WoT, C, C);

  qkv_gemm<<<dim3(M / 32, 3072 / 256), 128, 0, stream>>>(Xb, WqT, bqkv, Qb, Kb, Vt);
  attn_fa<<<dim3(64, 2048 / 16), 32, 0, stream>>>(Qb, Kb, Vt, Yb);
  out_gemm<<<dim3(M / 32, C / 256), 128, 0, stream>>>(Yb, WoT, bout, out);
}
